// RNNPolicy_38603166056811
// MI455X (gfx1250) — matrix operand order repaired and checked by reading, not compiled, not run on hardware
//
#include <hip/hip_runtime.h>
#include <cstdint>

// ---------------- types for WMMA ----------------
typedef __attribute__((ext_vector_type(16))) __bf16 v16bf;
typedef __attribute__((ext_vector_type(8)))  float  v8f;

union FragBF {
    uint4 q[2];
    v16bf v;
};

__device__ __forceinline__ uint16_t f2bf(float f) {
    uint32_t u = __float_as_uint(f);
    uint32_t r = u + 0x7FFFu + ((u >> 16) & 1u);   // round-to-nearest-even
    return (uint16_t)(r >> 16);
}
__device__ __forceinline__ uint32_t pack2bf(float a, float b) {
    return (uint32_t)f2bf(a) | ((uint32_t)f2bf(b) << 16);
}
__device__ __forceinline__ uint32_t hash3(uint32_t a, uint32_t b, uint32_t c) {
    uint32_t h = a * 0x9E3779B1u ^ b * 0x85EBCA77u ^ c * 0xC2B2AE3Du;
    h ^= h >> 16; h *= 0x7FEB352Du; h ^= h >> 15; h *= 0x846CA68Bu; h ^= h >> 16;
    return h;
}

// ---------------- problem dims ----------------
#define MB      4096
#define K1REAL  9225
#define K1P     9248      /* 289*32 */
#define D1REAL  3518
#define D1P     3520
#define D2REAL  1342
#define D2P     1344
#define LSTMN   512
#define GATES   2048
#define AZLD    544       /* 513 padded to 17*32 */

// tile counts (ktiles = Kp/32, ntiles = ceil(N/16))
#define KT1 289
#define NT1 220
#define KT2 110
#define NT2 84
#define KT3 42
#define NT3 32
#define KTL 16
#define NTL 128
#define KTD1 16
#define NTD1 16
#define KTH 8
#define NTH 4
#define KTO 17
#define NTO 32

// ---------------- workspace plan (bytes) ----------------
// phase1: A0 | W1P | X1     phase2: W2P,X2 overlay A0     phase3: W3P,X3 overlay W1P
// phase4: WihP/WhhP/H0P/BIF overlay A0 head, G1 overlays X2 (accum GEMM folds G2 into G1)
// phase5: AzA/AzB overlay phase-4 head, Az1/logits/decoder packs overlay dead G1
static const size_t OFF_A0   = 0;                 // 4096*9248*2      = 75,759,616
static const size_t OFF_W1P  = 75759616;          // 220*289*1024     = 65,105,920
static const size_t OFF_X1   = 140865536;         // 4096*3520*2     = 28,835,840
static const size_t OFF_W2P  = 0;
static const size_t OFF_X2   = 16777216;
static const size_t OFF_W3P  = 75759616;
static const size_t OFF_X3   = 77856768;
static const size_t OFF_WIHP = 0;
static const size_t OFF_WHHP = 2097152;
static const size_t OFF_H0P  = 4194304;
static const size_t OFF_BIF  = 8388608;
static const size_t OFF_G1   = 16777216;          // 4096*2048*4 = 33,554,432
static const size_t OFF_AZA  = 0;                 // 4096*544*2 = 4,456,448
static const size_t OFF_AZB  = 8388608;
static const size_t OFF_AZ1  = 16777216;
static const size_t OFF_LOG  = 20971520;
static const size_t OFF_DW1P = 25165824;
static const size_t OFF_DWOP = 26214400;
static const size_t OFF_HWP  = 27262976;          // 8 heads * 32768 B
static const size_t WS_REQ   = 169701376;

// =====================================================================
// Pack concatenated features to bf16, K padded with zeros to 9248.
// =====================================================================
__global__ void pack_feats(const float* __restrict__ a, const float* __restrict__ o,
                           const float* __restrict__ la, const float* __restrict__ ep,
                           uint16_t* __restrict__ dst)
{
    size_t idx = (size_t)blockIdx.x * blockDim.x + threadIdx.x;
    if (idx >= (size_t)MB * K1P) return;
    int row = (int)(idx / K1P);
    int col = (int)(idx % K1P);
    float v = 0.0f;
    if (col < 4608)        v = a[(size_t)row * 4608 + col];
    else if (col < 9216)   v = o[(size_t)row * 4608 + (col - 4608)];
    else if (col < 9224)   v = la[row * 8 + (col - 9216)];
    else if (col == 9224)  v = ep[0];
    dst[idx] = f2bf(v);
}

// =====================================================================
// Pack a weight matrix B[k][n] (optionally B = src^T) into bf16
// WMMA B-fragment order:  [ntile][ktile][lane][8 x u32]
// lane = 16*half + n%16 ; u32 v holds K = kt*32 + 8*half + 2v (+8 if v>=4) (+1)
// =====================================================================
__global__ void pack_weights(const float* __restrict__ src, int ldsrc, int K, int N,
                             int trans, uint32_t* __restrict__ dst, int ktiles,
                             long long total)
{
    long long idx = (long long)blockIdx.x * blockDim.x + threadIdx.x;
    if (idx >= total) return;
    int v    = (int)(idx & 7);
    int lane = (int)((idx >> 3) & 31);
    long long tile = idx >> 8;
    int kt = (int)(tile % ktiles);
    int nt = (int)(tile / ktiles);
    int half = lane >> 4;
    int n  = nt * 16 + (lane & 15);
    int k0 = kt * 32 + half * 8 + v * 2 + (v >= 4 ? 8 : 0);
    int k1 = k0 + 1;
    float f0 = 0.0f, f1 = 0.0f;
    if (n < N) {
        if (k0 < K) f0 = trans ? src[(size_t)n * ldsrc + k0] : src[(size_t)k0 * ldsrc + n];
        if (k1 < K) f1 = trans ? src[(size_t)n * ldsrc + k1] : src[(size_t)k1 * ldsrc + n];
    }
    dst[idx] = pack2bf(f0, f1);
}

// =====================================================================
// f32 -> bf16 row-major copy (for h0)
// =====================================================================
__global__ void conv_bf16(const float* __restrict__ src, uint16_t* __restrict__ dst, long long n)
{
    long long idx = (long long)blockIdx.x * blockDim.x + threadIdx.x;
    if (idx < n) dst[idx] = f2bf(src[idx]);
}

__global__ void add_bias2(const float* __restrict__ a, const float* __restrict__ b,
                          float* __restrict__ dst, int n)
{
    int i = blockIdx.x * blockDim.x + threadIdx.x;
    if (i < n) dst[i] = a[i] + b[i];
}

// =====================================================================
// bf16 WMMA GEMM: C = act( A[M,Kp] @ Bpacked + bias (+ C) )
// 256 threads = 8 waves arranged 4(M) x 2(N); wave tile 32x64
//   -> block tile 128x128, 8 WMMAs per 12 b128 loads per K-step.
// Out-of-range N tiles are handled by CLAMPING the B tile index (loads
// duplicate valid data, results never stored) -> no exec-mask churn in
// the inner loop, keeps EXEC all-ones for WMMA.
// A: bf16 row-major, lda elements (mult of 8). Bp: packed fragments.
// Output: f32 (Cf) or bf16 (Cb); columns >= Nreal stored as 0.
// =====================================================================
__global__ __launch_bounds__(256) void gemm_bf16_wmma(
    const uint16_t* __restrict__ A, int lda,
    const uint32_t* __restrict__ Bp, int ktiles, int ntilesB,
    const float* __restrict__ bias,
    float* __restrict__ Cf, uint16_t* __restrict__ Cb,
    int ldc, int Nstore, int Nreal, int doRelu, int accumC)
{
    const int lane = threadIdx.x & 31;
    const int wave = threadIdx.x >> 5;
    const int m0 = blockIdx.x * 128 + (wave & 3) * 32;   // 4 waves along M
    const int n0 = blockIdx.y * 128 + (wave >> 2) * 64;  // 2 waves along N
    const int half = lane >> 4;
    const int lan  = lane & 15;

    v8f acc[2][4];
#pragma unroll
    for (int i = 0; i < 2; ++i)
#pragma unroll
        for (int j = 0; j < 4; ++j)
#pragma unroll
            for (int e = 0; e < 8; ++e) acc[i][j][e] = 0.0f;

    const uint16_t* aRow0 = A + (size_t)(m0 + lan) * lda + half * 8;
    const uint16_t* aRow1 = aRow0 + (size_t)16 * lda;

    const uint32_t* bBase[4];
#pragma unroll
    for (int j = 0; j < 4; ++j) {
        int nt = (n0 >> 4) + j;
        if (nt > ntilesB - 1) nt = ntilesB - 1;          // clamp: always-valid loads
        bBase[j] = Bp + ((size_t)nt * ktiles * 32 + lane) * 8;
    }

    for (int kt = 0; kt < ktiles; ++kt) {
        FragBF a0, a1, b[4];
        const uint16_t* ap0 = aRow0 + kt * 32;
        a0.q[0] = *(const uint4*)(ap0);
        a0.q[1] = *(const uint4*)(ap0 + 16);
        const uint16_t* ap1 = aRow1 + kt * 32;
        a1.q[0] = *(const uint4*)(ap1);
        a1.q[1] = *(const uint4*)(ap1 + 16);
#pragma unroll
        for (int j = 0; j < 4; ++j) {
            const uint32_t* bp = bBase[j] + (size_t)kt * 256;
            b[j].q[0] = *(const uint4*)(bp);
            b[j].q[1] = *(const uint4*)(bp + 4);
        }
#pragma unroll
        for (int j = 0; j < 4; ++j) {
            acc[0][j] = __builtin_amdgcn_wmma_f32_16x16x32_bf16(false, a0.v, false, b[j].v,
                                                                (short)0, acc[0][j], false, false);
            acc[1][j] = __builtin_amdgcn_wmma_f32_16x16x32_bf16(false, a1.v, false, b[j].v,
                                                                (short)0, acc[1][j], false, false);
        }
    }

    // epilogue: C layout -> (m = base + 8*half + r, n = n0 + 16*tj + lane%16)
#pragma unroll
    for (int ti = 0; ti < 2; ++ti) {
#pragma unroll
        for (int tj = 0; tj < 4; ++tj) {
            int n = n0 + tj * 16 + lan;
            if (n >= Nstore) continue;
            float bv = (bias != nullptr && n < Nreal) ? bias[n] : 0.0f;
            int mbase = m0 + ti * 16 + half * 8;
#pragma unroll
            for (int r = 0; r < 8; ++r) {
                size_t off = (size_t)(mbase + r) * ldc + n;
                float v = acc[ti][tj][r] + bv;
                if (accumC) v += Cf[off];
                if (doRelu) v = fmaxf(v, 0.0f);
                if (n >= Nreal) v = 0.0f;
                if (Cf) Cf[off] = v;
                else    Cb[off] = f2bf(v);
            }
        }
    }
}

// =====================================================================
// LSTM cell (h0 contribution already accumulated into gates buffer)
// gate order i,f,g,o ; writes z as bf16 into Az (ld=544)
// =====================================================================
__global__ void lstm_cell(const float* __restrict__ gates, const float* __restrict__ c0,
                          uint16_t* __restrict__ Az)
{
    int idx = blockIdx.x * blockDim.x + threadIdx.x;
    if (idx >= MB * LSTMN) return;
    int b = idx >> 9;
    int j = idx & 511;
    const float* row = gates + (size_t)b * GATES;
    float gi = row[j], gf = row[512 + j], gg = row[1024 + j], go = row[1536 + j];
    float si = 1.0f / (1.0f + expf(-gi));
    float sf = 1.0f / (1.0f + expf(-gf));
    float so = 1.0f / (1.0f + expf(-go));
    float c = sf * c0[idx] + si * tanhf(gg);
    float z = so * tanhf(c);
    Az[(size_t)b * AZLD + j] = f2bf(z);
}

// =====================================================================
// Head: log-softmax over 64 logits, hash-Gumbel categorical sample,
// entropy + logprob accumulation, write action (f32 out + bf16 col 512)
// out layout: [0,32768) action[b][t] ; 32768 entropy ; [32769,36865) logp
// =====================================================================
__global__ void head_sample(const float* __restrict__ logits, int t,
                            float* __restrict__ out, uint16_t* __restrict__ AzCur)
{
    __shared__ float red[256];
    int row = blockIdx.x * 256 + threadIdx.x;
    float ent = 0.0f;
    if (row < MB) {
        const float* l = logits + (size_t)row * 64;
        float mx = -1e30f;
        for (int j = 0; j < 64; ++j) mx = fmaxf(mx, l[j]);
        float s = 0.0f;
        for (int j = 0; j < 64; ++j) s += expf(l[j] - mx);
        float lse = mx + logf(s);
        int best = 0;
        float bestv = -1e30f;
        for (int j = 0; j < 64; ++j) {
            float lp = l[j] - lse;
            ent -= expf(lp) * lp;
            uint32_t h = hash3((uint32_t)row, (uint32_t)t, (uint32_t)j);
            float u = fmaxf((float)(h >> 8) * (1.0f / 16777216.0f), 1e-12f);
            float g = -logf(fmaxf(-logf(u), 1e-12f));
            float v = l[j] + g;
            if (v > bestv) { bestv = v; best = j; }
        }
        out[(size_t)row * 8 + t] = (float)best;
        out[32769 + row] += l[best] - lse;
        AzCur[(size_t)row * AZLD + 512] = f2bf((float)best);
    }
    red[threadIdx.x] = ent;
    __syncthreads();
    for (int s2 = 128; s2 > 0; s2 >>= 1) {
        if (threadIdx.x < s2) red[threadIdx.x] += red[threadIdx.x + s2];
        __syncthreads();
    }
    if (threadIdx.x == 0) atomicAdd(out + 32768, red[0]);
}

// =====================================================================
// host launch
// =====================================================================
extern "C" void kernel_launch(void* const* d_in, const int* in_sizes, int n_in,
                              void* d_out, int out_size, void* d_ws, size_t ws_size,
                              hipStream_t stream)
{
    (void)in_sizes; (void)out_size;
    if (ws_size < WS_REQ) return;
    char* ws = (char*)d_ws;
    float* out = (float*)d_out;

    const float* actual = (const float*)d_in[0];
    const float* object = (const float*)d_in[1];
    const float* lastac = (const float*)d_in[2];
    const float* ep     = (const float*)d_in[3];
    const float* W1  = (const float*)d_in[4];
    const float* b1  = (const float*)d_in[5];
    const float* W2  = (const float*)d_in[6];
    const float* b2  = (const float*)d_in[7];
    const float* W3  = (const float*)d_in[8];
    const float* b3  = (const float*)d_in[9];
    const float* Wih = (const float*)d_in[10];
    const float* Whh = (const float*)d_in[11];
    const float* bih = (const float*)d_in[12];
    const float* bhh = (const float*)d_in[13];
    const float* h0  = (const float*)d_in[14];
    const float* c0  = (const float*)d_in[15];
    const float* dW1 = (const float*)d_in[16];
    const float* db1 = (const float*)d_in[17];
    const bool flat = (n_in >= 36);   // tuples expanded vs packed
    const float* dWo = flat ? (const float*)d_in[34] : (const float*)d_in[20];
    const float* dbo = flat ? (const float*)d_in[35] : (const float*)d_in[21];
    auto headW = [&](int t) -> const float* {
        return flat ? (const float*)d_in[18 + t] : ((const float*)d_in[18]) + (size_t)t * 256 * 64;
    };
    auto headb = [&](int t) -> const float* {
        return flat ? (const float*)d_in[26 + t] : ((const float*)d_in[19]) + (size_t)t * 64;
    };

    uint16_t* A0   = (uint16_t*)(ws + OFF_A0);
    uint32_t* W1P  = (uint32_t*)(ws + OFF_W1P);
    uint16_t* X1   = (uint16_t*)(ws + OFF_X1);
    uint32_t* W2P  = (uint32_t*)(ws + OFF_W2P);
    uint16_t* X2   = (uint16_t*)(ws + OFF_X2);
    uint32_t* W3P  = (uint32_t*)(ws + OFF_W3P);
    uint16_t* X3   = (uint16_t*)(ws + OFF_X3);
    uint32_t* WIHP = (uint32_t*)(ws + OFF_WIHP);
    uint32_t* WHHP = (uint32_t*)(ws + OFF_WHHP);
    uint16_t* H0P  = (uint16_t*)(ws + OFF_H0P);
    float*    BIF  = (float*)(ws + OFF_BIF);
    float*    G1   = (float*)(ws + OFF_G1);
    uint16_t* AZA  = (uint16_t*)(ws + OFF_AZA);
    uint16_t* AZB  = (uint16_t*)(ws + OFF_AZB);
    uint16_t* AZ1  = (uint16_t*)(ws + OFF_AZ1);
    float*    LOGI = (float*)(ws + OFF_LOG);
    uint32_t* DW1P = (uint32_t*)(ws + OFF_DW1P);
    uint32_t* DWOP = (uint32_t*)(ws + OFF_DWOP);
    uint32_t* HWP  = (uint32_t*)(ws + OFF_HWP);

    auto launch_pack = [&](const float* src, int lds, int K, int N, int trans,
                           uint32_t* dst, int ktiles, int ntiles) {
        long long total = (long long)ntiles * ktiles * 256;
        int blocks = (int)((total + 255) / 256);
        pack_weights<<<blocks, 256, 0, stream>>>(src, lds, K, N, trans, dst, ktiles, total);
    };
    auto launch_gemm = [&](const uint16_t* A, int lda, const uint32_t* Bp, int ktiles,
                           int ntilesB, const float* bias, float* Cf, uint16_t* Cb,
                           int ldc, int Nstore, int Nreal, int relu, int accum) {
        dim3 grid(MB / 128, (ntilesB + 7) / 8);
        gemm_bf16_wmma<<<grid, 256, 0, stream>>>(A, lda, Bp, ktiles, ntilesB, bias,
                                                 Cf, Cb, ldc, Nstore, Nreal, relu, accum);
    };

    // zero entropy + log-probability accumulators in d_out
    hipMemsetAsync((char*)d_out + (size_t)32768 * sizeof(float), 0,
                   (size_t)4097 * sizeof(float), stream);

    // ---- encoder ----
    {
        long long total = (long long)MB * K1P;
        pack_feats<<<(int)((total + 255) / 256), 256, 0, stream>>>(actual, object, lastac, ep, A0);
    }
    launch_pack(W1, D1REAL, K1REAL, D1REAL, 0, W1P, KT1, NT1);
    launch_gemm(A0, K1P, W1P, KT1, NT1, b1, nullptr, X1, D1P, D1P, D1REAL, 1, 0);

    launch_pack(W2, D2REAL, D1REAL, D2REAL, 0, W2P, KT2, NT2);   // overlays dead A0
    launch_gemm(X1, D1P, W2P, KT2, NT2, b2, nullptr, X2, D2P, D2P, D2REAL, 1, 0);

    launch_pack(W3, LSTMN, D2REAL, LSTMN, 0, W3P, KT3, NT3);     // overlays dead W1P
    launch_gemm(X2, D2P, W3P, KT3, NT3, b3, nullptr, X3, LSTMN, LSTMN, LSTMN, 1, 0);

    // ---- LSTM gates: G1 = X3 @ Wih^T + (bih+bhh) ; G1 += h0 @ Whh^T ----
    launch_pack(Wih, LSTMN, LSTMN, GATES, 1, WIHP, KTL, NTL);
    launch_pack(Whh, LSTMN, LSTMN, GATES, 1, WHHP, KTL, NTL);
    conv_bf16<<<(MB * LSTMN + 255) / 256, 256, 0, stream>>>(h0, H0P, (long long)MB * LSTMN);
    add_bias2<<<(GATES + 255) / 256, 256, 0, stream>>>(bih, bhh, BIF, GATES);
    launch_gemm(X3, LSTMN, WIHP, KTL, NTL, BIF, G1, nullptr, GATES, GATES, GATES, 0, 0);
    launch_gemm(H0P, LSTMN, WHHP, KTL, NTL, nullptr, G1, nullptr, GATES, GATES, GATES, 0, 1);

    // ---- z seed (Az buffers zeroed: cols 512..543 must start at 0) ----
    hipMemsetAsync(AZA, 0, (size_t)MB * AZLD * 2, stream);
    hipMemsetAsync(AZB, 0, (size_t)MB * AZLD * 2, stream);
    lstm_cell<<<(MB * LSTMN + 255) / 256, 256, 0, stream>>>(G1, c0, AZA);

    // ---- decoder weight packs (overlay dead G1 region; ordered after lstm) ----
    launch_pack(dW1, 256, LSTMN, 256, 0, DW1P, KTD1, NTD1);
    launch_pack(dWo, LSTMN, LSTMN + 1, LSTMN, 0, DWOP, KTO, NTO);
    for (int t = 0; t < 8; ++t)
        launch_pack(headW(t), 64, 256, 64, 0, HWP + (size_t)t * 8192, KTH, NTH);

    // ---- autoregressive decoder ----
    uint16_t* AzCur = AZA;
    uint16_t* AzNxt = AZB;
    for (int t = 0; t < 8; ++t) {
        // z1 = relu(z @ dW1 + db1)
        launch_gemm(AzCur, AZLD, DW1P, KTD1, NTD1, db1, nullptr, AZ1, 256, 256, 256, 1, 0);
        // logits = z1 @ Wk + bk
        launch_gemm(AZ1, 256, HWP + (size_t)t * 8192, KTH, NTH, headb(t),
                    LOGI, nullptr, 64, 64, 64, 0, 0);
        // sample + bookkeeping + write action into col 512 of current z buffer
        head_sample<<<MB / 256, 256, 0, stream>>>(LOGI, t, out, AzCur);
        if (t < 7) {
            // z = relu(concat([z, act]) @ dWo + dbo)   (K=513 padded to 544)
            launch_gemm(AzCur, AZLD, DWOP, KTO, NTO, dbo, nullptr, AzNxt, AZLD,
                        LSTMN, LSTMN, 1, 0);
            uint16_t* tmp = AzCur; AzCur = AzNxt; AzNxt = tmp;
        }
    }
}
